// AngularSoftmax_41351945126507
// MI455X (gfx1250) — matrix operand order repaired and checked by reading, not compiled, not run on hardware
//
#include <hip/hip_runtime.h>
#include <hip/hip_bf16.h>

#define BATCH   256
#define FEAT    512
#define CLASSES 100000
#define NTILES  (CLASSES / 16)   // 6250 column tiles
#define EPS     1e-6f

typedef __attribute__((ext_vector_type(16))) __bf16 v16bf;
typedef __attribute__((ext_vector_type(8)))  float  v8f;

// float -> bf16 bits, round-to-nearest-even
__device__ __forceinline__ unsigned short f2bf(float f) {
    unsigned int u = __float_as_uint(f);
    u = (u + 0x7FFFu + ((u >> 16) & 1u)) >> 16;
    return (unsigned short)u;
}

__device__ __forceinline__ float block_reduce_256(float s, float* sm) {
    // 256 threads = 8 waves of 32
    for (int off = 16; off; off >>= 1) s += __shfl_xor(s, off, 32);
    if ((threadIdx.x & 31) == 0) sm[threadIdx.x >> 5] = s;
    __syncthreads();
    float v = (threadIdx.x < 8) ? sm[threadIdx.x] : 0.0f;
    for (int off = 4; off; off >>= 1) v += __shfl_xor(v, off, 32);
    return v; // valid in thread 0
}

// ---- K1: rn[f] = 1/||W[f,:]||  (512 blocks x 256 threads) ----
__global__ __launch_bounds__(256) void k_wnorm(const float* __restrict__ W,
                                               float* __restrict__ rn) {
    __shared__ float sm[8];
    const int f = blockIdx.x;
    const float4* p = (const float4*)(W + (size_t)f * CLASSES);
    float s = 0.0f;
    for (int i = threadIdx.x; i < CLASSES / 4; i += 256) {
        float4 v = p[i];
        s += v.x * v.x + v.y * v.y + v.z * v.z + v.w * v.w;
    }
    float tot = block_reduce_256(s, sm);
    if (threadIdx.x == 0) rn[f] = 1.0f / sqrtf(tot);
}

// ---- K2: mod_x[b] = ||x[b,:]||  (256 blocks x 256 threads) ----
__global__ __launch_bounds__(256) void k_xnorm(const float* __restrict__ x,
                                               float* __restrict__ modx) {
    __shared__ float sm[8];
    const int b = blockIdx.x;
    const float* p = x + (size_t)b * FEAT;
    float s = 0.0f;
    for (int i = threadIdx.x; i < FEAT; i += 256) { float v = p[i]; s += v * v; }
    float tot = block_reduce_256(s, sm);
    if (threadIdx.x == 0) modx[b] = sqrtf(tot);
}

// ---- K3: A_bf16[b,f] = bf16(x[b,f] * rn[f])  (512 blocks x 256) ----
__global__ __launch_bounds__(256) void k_mkA(const float* __restrict__ x,
                                             const float* __restrict__ rn,
                                             unsigned short* __restrict__ A) {
    int idx = blockIdx.x * 256 + threadIdx.x;   // 0 .. 256*512-1
    int f = idx & (FEAT - 1);
    A[idx] = f2bf(x[idx] * rn[f]);
}

// ---- K4: WMMA GEMM xw = A @ W, epilogue writes xw + per-tile e_cos row sums ----
// grid = 6250 column tiles, block = 512 threads = 16 waves (one 16x16 M-tile each)
__global__ __launch_bounds__(512) void k_gemm(const float* __restrict__ W,
                                              const unsigned short* __restrict__ A,
                                              const float* __restrict__ modx,
                                              float* __restrict__ out,
                                              float* __restrict__ partials) {
    __shared__ unsigned short Blds[16][FEAT + 8];   // [col][k], +16B pad
    const int c0 = blockIdx.x * 16;
    const int t  = threadIdx.x;

    // Stage W tile [512 K x 16 cols] -> bf16, transposed into LDS (col-major in K)
    for (int it = 0; it < 4; ++it) {
        int f  = it * 128 + (t >> 2);
        int jj = (t & 3) * 4;
        float4 v = *(const float4*)(W + (size_t)f * CLASSES + c0 + jj);
        Blds[jj + 0][f] = f2bf(v.x);
        Blds[jj + 1][f] = f2bf(v.y);
        Blds[jj + 2][f] = f2bf(v.z);
        Blds[jj + 3][f] = f2bf(v.w);
    }
    __syncthreads();

    const int wave = t >> 5;          // 0..15 -> M tile
    const int lane = t & 31;
    const int lh   = lane >> 4;       // half of wave
    const int ln   = lane & 15;

    union { v8f v; float f[8]; } acc;
#pragma unroll
    for (int i = 0; i < 8; ++i) acc.f[i] = 0.0f;

    const unsigned short* arow = A + (size_t)(wave * 16 + ln) * FEAT;

    for (int kk = 0; kk < FEAT / 32; ++kk) {
        union { v16bf v; uint4 u[2]; } af, bfr;
        // A fragment (16-bit A 16x32 layout): lane half picks K base; two 16B segs
        int ka = kk * 32 + lh * 8;
        af.u[0] = *(const uint4*)(arow + ka);
        af.u[1] = *(const uint4*)(arow + ka + 16);
        // B fragment (16-bit B 32x16 layout): lane half picks K base; two 16B segs (same K order as A)
        int kb = kk * 32 + lh * 8;
        bfr.u[0] = *(const uint4*)(&Blds[ln][kb]);
        bfr.u[1] = *(const uint4*)(&Blds[ln][kb + 16]);
        acc.v = __builtin_amdgcn_wmma_f32_16x16x32_bf16(
            false, af.v, false, bfr.v, (short)0, acc.v, false, false);
    }

    // Epilogue: VGPR r holds row (wave*16 + r + 8*lh), col = c0 + ln
#pragma unroll
    for (int r = 0; r < 8; ++r) {
        int row = wave * 16 + r + 8 * lh;
        float xw = acc.f[r];
        out[(size_t)row * CLASSES + c0 + ln] = xw;
        float mx = modx[row];
        float e  = expf(xw * (mx / (mx + EPS)));      // e_cos
        for (int off = 1; off < 16; off <<= 1) e += __shfl_xor(e, off, 32);
        if (ln == 0) partials[(size_t)blockIdx.x * BATCH + row] = e;
    }
}

// ---- K5: total[b] = sum over tiles of partials  (256 blocks x 256) ----
__global__ __launch_bounds__(256) void k_totals(const float* __restrict__ partials,
                                                float* __restrict__ totals) {
    __shared__ float sm[8];
    const int row = blockIdx.x;
    float s = 0.0f;
    for (int j = threadIdx.x; j < NTILES; j += 256)
        s += partials[(size_t)j * BATCH + row];
    float tot = block_reduce_256(s, sm);
    if (threadIdx.x == 0) totals[row] = tot;
}

// ---- K6: finalize out = log(e_psi / (e_psi + total - e_cos)) in place ----
__global__ __launch_bounds__(256) void k_finalize(float* __restrict__ out,
                                                  const float* __restrict__ modx,
                                                  const float* __restrict__ totals) {
    unsigned int i4 = blockIdx.x * 256 + threadIdx.x;   // float4 index
    float4* p = (float4*)out;
    float4 v = p[i4];
    unsigned int row = (i4 * 4u) / (unsigned int)CLASSES;  // rows are %4 aligned
    float mx  = modx[row];
    float T   = totals[row];
    float inv = 1.0f / (mx + EPS);
    float o[4] = {v.x, v.y, v.z, v.w};
#pragma unroll
    for (int j = 0; j < 4; ++j) {
        float xw  = o[j];
        float c   = xw * inv;              // cos_theta
        float ec  = expf(xw * mx * inv);   // e_cos = exp(mod_x * cos_theta)
        float c2  = c * c;
        float psi = -(8.0f * c2 * c2 - 8.0f * c2 + 1.0f) - 6.0f;  // k=3
        float ep  = expf(mx * psi);        // e_psi (underflows like reference)
        o[j] = logf(ep / (ep + T - ec));   // literal reference formula
    }
    p[i4] = make_float4(o[0], o[1], o[2], o[3]);
}

extern "C" void kernel_launch(void* const* d_in, const int* in_sizes, int n_in,
                              void* d_out, int out_size, void* d_ws, size_t ws_size,
                              hipStream_t stream) {
    const float* x = (const float*)d_in[0];   // [256, 512]
    const float* W = (const float*)d_in[1];   // [512, 100000]
    float* out = (float*)d_out;               // [256, 100000]

    char* ws = (char*)d_ws;
    float*          rn       = (float*)(ws);                      // 512 f
    float*          modx     = (float*)(ws + 4096);               // 256 f
    unsigned short* A        = (unsigned short*)(ws + 8192);      // 256*512 bf16 = 256KB
    float*          partials = (float*)(ws + 8192 + 262144);      // 6250*256 f = 6.4MB
    float*          totals   = (float*)(ws + 8192 + 262144 + (size_t)NTILES * BATCH * 4);

    k_wnorm   <<<FEAT,  256, 0, stream>>>(W, rn);
    k_xnorm   <<<BATCH, 256, 0, stream>>>(x, modx);
    k_mkA     <<<(BATCH * FEAT) / 256, 256, 0, stream>>>(x, rn, A);
    k_gemm    <<<NTILES, 512, 0, stream>>>(W, A, modx, out, partials);
    k_totals  <<<BATCH, 256, 0, stream>>>(partials, totals);
    k_finalize<<<(BATCH * CLASSES) / 4 / 256, 256, 0, stream>>>(out, modx, totals);
}
